// DeepseekMLA_42262478193005
// MI455X (gfx1250) — matrix operand order repaired and checked by reading, not compiled, not run on hardware
//
#include <hip/hip_runtime.h>
#include <hip/hip_bf16.h>

// CDNA5 / gfx1250 wave32 WMMA implementation of the DeepSeek MLA indexer.
// Pipeline (all on `stream`):
//   1) mla_qgemm_rope : q = qr @ wq_b (bf16 WMMA, f32 acc) + fused interleaved RoPE,
//                       emitted as bf16 [4096][64*128]
//   2) mla_kwprep     : k = rope(layernorm(x @ wk)) -> bf16 [4096][128]; w = x @ w_weights
//   3) mla_score      : per (128q x 128k) tile: per-head 16x16x32 bf16 WMMA chain (K=128),
//                       relu, weighted head-combine, scale, causal mask -> f32 scores.
//                       K tile staged to LDS with global_load_async_to_lds_b128 (ASYNCcnt).
//   4) mla_topk       : per-row bitonic sort of 4096 (val,idx), write top 2048
// Workspace: 130 MB (q_bf16 64MB @0, k_bf16 1MB @64MB, w 1MB @65MB, scores 64MB @66MB).

typedef __attribute__((ext_vector_type(16))) __bf16 v16bf;
typedef __attribute__((ext_vector_type(8)))  __bf16 v8bf;
typedef __attribute__((ext_vector_type(8)))  float  v8f;
typedef __attribute__((ext_vector_type(4)))  float  v4f;

union ABf { v16bf v; v8bf h[2]; };

#define T_TOK   4096
#define DMODEL  7168
#define QRDIM   1536
#define NH      64
#define HD      128
#define QCOLS   (NH * HD)   /* 8192 */
#define TOPK_N  2048
#define SCALE_F 0.08838834764831845f /* 128^-0.5 */

static __device__ __forceinline__ v8f wmma_bf16(v16bf a, v16bf b, v8f c) {
  // D = A(16x32 bf16) x B(32x16 bf16) + C(16x16 f32)
  return __builtin_amdgcn_wmma_f32_16x16x32_bf16(false, a, false, b, (short)0, c,
                                                 false, false);
}

static __device__ __forceinline__ unsigned pack_bf16(float a, float b) {
  union { __bf16 h[2]; unsigned u; } p;
  p.h[0] = (__bf16)a;
  p.h[1] = (__bf16)b;
  return p.u;  // lowers to v_cvt_pk_bf16_f32
}

// ---------------------------------------------------------------------------
// Kernel 1: q = qr @ wq_b with fused interleaved RoPE, output bf16.
// Grid (QCOLS/128, T/128), 256 threads = 8 waves. Wave wv owns rows wv*16..+15
// of the 128x128 C tile (8 ni sub-tiles, 64 f32 acc VGPRs).
// ---------------------------------------------------------------------------
__global__ __launch_bounds__(256) void mla_qgemm_rope(
    const float* __restrict__ qr, const float* __restrict__ wqb,
    const float* __restrict__ cosp, const float* __restrict__ sinp,
    __bf16* __restrict__ qb) {
  __shared__ __align__(16) __bf16 sA[128 * 32];  // [row][k] row-major
  __shared__ __align__(16) __bf16 sB[128 * 40];  // [col][k] transposed, padded

  const int t = threadIdx.x;
  const int lane = t & 31, wv = t >> 5;
  const int m0 = blockIdx.y * 128, n0 = blockIdx.x * 128;

  v8f acc[8];
  for (int i = 0; i < 8; ++i)
    for (int j = 0; j < 8; ++j) acc[i][j] = 0.f;

  const int m   = lane & 15;
  const int lhA = (lane < 16) ? 0 : 8;   // A: K start within first half
  const int d0B = (lane < 16) ? 0 : 8;   // B: K start within first half

  for (int k0 = 0; k0 < QRDIM; k0 += 32) {
    // stage A: 128x32, two f32 -> packed bf16x2 b32 stores
    for (int i = 0; i < 8; ++i) {
      int idx = t + 256 * i;        // 0..2047
      int row = idx >> 4;
      int c2  = (idx & 15) * 2;
      const float* qp = &qr[(size_t)(m0 + row) * QRDIM + k0 + c2];
      *(unsigned*)&sA[row * 32 + c2] = pack_bf16(qp[0], qp[1]);
    }
    // stage B transposed: two k-rows packed per b32 store
    for (int i = 0; i < 8; ++i) {
      int idx = t + 256 * i;        // 0..2047
      int col = idx & 127;
      int r2  = (idx >> 7) * 2;
      float f0 = wqb[(size_t)(k0 + r2) * QCOLS + n0 + col];
      float f1 = wqb[(size_t)(k0 + r2 + 1) * QCOLS + n0 + col];
      *(unsigned*)&sB[col * 40 + r2] = pack_bf16(f0, f1);
    }
    __syncthreads();

    ABf af;
    int arow = wv * 16 + m;
    af.h[0] = *(const v8bf*)&sA[arow * 32 + lhA];
    af.h[1] = *(const v8bf*)&sA[arow * 32 + lhA + 16];
#pragma unroll
    for (int ni = 0; ni < 8; ++ni) {
      ABf bf_;
      int n = ni * 16 + m;
      bf_.h[0] = *(const v8bf*)&sB[n * 40 + d0B];
      bf_.h[1] = *(const v8bf*)&sB[n * 40 + d0B + 16];
      acc[ni] = wmma_bf16(af.v, bf_.v, acc[ni]);
    }
    __syncthreads();
  }

  // Epilogue: fused RoPE on head-dims [0,64), bf16 store.
  const int rj = (lane < 16) ? 0 : 8;
  for (int ni = 0; ni < 8; ++ni) {
    int cb = n0 + ni * 16;
    int dbase = cb & 127;          // dim within head (tile-uniform)
    int col = cb + m;
    bool rope = (dbase < 64);
#pragma unroll
    for (int j = 0; j < 8; ++j) {
      int row = m0 + wv * 16 + j + rj;
      float v = acc[ni][j];
      float other = __shfl_xor(v, 1, 32);  // pair column lives in lane^1
      if (rope) {
        int d = dbase + m;
        float c = cosp[row * 32 + (d >> 1)];
        float s = sinp[row * 32 + (d >> 1)];
        v = (d & 1) ? (v * c + other * s) : (v * c - other * s);
      }
      qb[(size_t)row * QCOLS + col] = (__bf16)v;
    }
  }
}

// ---------------------------------------------------------------------------
// Kernel 2: k = rope(layernorm(x @ wk)) -> bf16; w = x @ w_weights -> f32.
// One block per token row; x row staged in LDS (28 KB of the 320 KB/WGP).
// ---------------------------------------------------------------------------
__global__ __launch_bounds__(256) void mla_kwprep(
    const float* __restrict__ x, const float* __restrict__ wk,
    const float* __restrict__ ww, const float* __restrict__ knw,
    const float* __restrict__ knb, const float* __restrict__ cosp,
    const float* __restrict__ sinp, __bf16* __restrict__ kb,
    float* __restrict__ wbuf) {
  __shared__ float xr[DMODEL];
  __shared__ float kv[128];
  __shared__ float red[128];
  __shared__ float knorm[128];
  const int t = threadIdx.x, r = blockIdx.x;

  for (int i = 0; i < 28; ++i)
    xr[t + 256 * i] = x[(size_t)r * DMODEL + t + 256 * i];
  __syncthreads();

  float kacc = 0.f;
  if (t < 128) {
#pragma unroll 4
    for (int d = 0; d < DMODEL; ++d) kacc += xr[d] * wk[(size_t)d * HD + t];
  } else if (t < 192) {
    int c = t - 128;
    float wa = 0.f;
#pragma unroll 4
    for (int d = 0; d < DMODEL; ++d) wa += xr[d] * ww[(size_t)d * NH + c];
    wbuf[(size_t)r * NH + c] = wa;
  }
  if (t < 128) { kv[t] = kacc; red[t] = kacc; }
  __syncthreads();
  for (int s2 = 64; s2 > 0; s2 >>= 1) {
    if (t < s2) red[t] += red[t + s2];
    __syncthreads();
  }
  float mu = red[0] * (1.f / 128.f);
  __syncthreads();
  if (t < 128) { float d = kv[t] - mu; red[t] = d * d; }
  __syncthreads();
  for (int s2 = 64; s2 > 0; s2 >>= 1) {
    if (t < s2) red[t] += red[t + s2];
    __syncthreads();
  }
  float var = red[0] * (1.f / 128.f);
  float rs = rsqrtf(var + 1e-6f);
  if (t < 128) knorm[t] = (kv[t] - mu) * rs * knw[t] + knb[t];
  __syncthreads();
  if (t < 32) {
    float x0 = knorm[2 * t], x1 = knorm[2 * t + 1];
    float c = cosp[r * 32 + t], s = sinp[r * 32 + t];
    kb[(size_t)r * HD + 2 * t]     = (__bf16)(x0 * c - x1 * s);
    kb[(size_t)r * HD + 2 * t + 1] = (__bf16)(x1 * c + x0 * s);
  } else if (t < 64) {
    int d = 2 * t;  // 64..126
    kb[(size_t)r * HD + d]     = (__bf16)knorm[d];
    kb[(size_t)r * HD + d + 1] = (__bf16)knorm[d + 1];
  }
}

// ---------------------------------------------------------------------------
// Kernel 3: combined causal scores. Grid (32 keytiles, 32 qchunks), 8 waves.
// LDS: K tile 128x128 bf16 (padded stride 136, 34 KB; staged with
// global_load_async_to_lds_b128) + w^T 64x128 f32 (32 KB).
// Two passes of 4 key sub-tiles: B fragments hoisted to 128 VGPRs, heads inner:
// per head 8 global b128 A loads feed 16 WMMAs, then relu + weighted combine
// with two broadcast ds_load_b128 of the per-head weight row.
// ---------------------------------------------------------------------------
__global__ __launch_bounds__(256) void mla_score(
    const __bf16* __restrict__ qb, const __bf16* __restrict__ kbm,
    const float* __restrict__ wbuf, float* __restrict__ scores) {
  __shared__ __align__(16) __bf16 kt[128 * 136];
  __shared__ __align__(16) float wt[64 * 128];
  const int t = threadIdx.x, lane = t & 31, wv = t >> 5;
  const int kbase = blockIdx.x * 128, qbase = blockIdx.y * 128;

  // Async-stage K tile: 2048 x 16B chunks, LDS dest padded to stride 272B.
  {
    unsigned ktbase = (unsigned)(uintptr_t)(&kt[0]);  // low 32 bits = LDS offset
    unsigned long long gk =
        (unsigned long long)(uintptr_t)(kbm + (size_t)kbase * HD);
    for (int i = 0; i < 8; ++i) {
      int idx = t + 256 * i;   // 0..2047
      int key = idx >> 4;      // 0..127
      int dc  = (idx & 15) * 16;
      unsigned lds  = ktbase + key * 272 + dc;
      unsigned voff = (unsigned)(key * 256 + dc);
      asm volatile("global_load_async_to_lds_b128 %0, %1, %2"
                   :: "v"(lds), "v"(voff), "s"(gk) : "memory");
    }
  }
  // Stage w^T (broadcast-friendly rows of 128 floats per head)
  {
    int h = t & 63, q2 = t >> 6;
    for (int i = 0; i < 32; ++i) {
      int ql = q2 + 4 * i;
      wt[h * 128 + ql] = wbuf[(size_t)(qbase + ql) * NH + h];
    }
  }
  asm volatile("s_wait_asynccnt 0x0" ::: "memory");
  __syncthreads();

  const int m  = lane & 15;
  const int rj = (lane < 16) ? 0 : 8;
  const int dA = (lane < 16) ? 0 : 8;
  const int dB = (lane < 16) ? 0 : 8;
  const __bf16* qrow = qb + (size_t)(qbase + wv * 16 + m) * QCOLS;

  for (int npass = 0; npass < 2; ++npass) {
    // Hoist B fragments for 4 key sub-tiles (4 ni x 4 ks x 8 VGPR = 128 VGPR)
    ABf bfr[4][4];
#pragma unroll
    for (int nil = 0; nil < 4; ++nil) {
      int key = (npass * 4 + nil) * 16 + m;
#pragma unroll
      for (int ks = 0; ks < 4; ++ks) {
        const __bf16* kp = &kt[key * 136 + ks * 32 + dB];
        bfr[nil][ks].h[0] = *(const v8bf*)kp;
        bfr[nil][ks].h[1] = *(const v8bf*)(kp + 16);
      }
    }
    v8f acc[4];
    for (int nil = 0; nil < 4; ++nil)
      for (int j = 0; j < 8; ++j) acc[nil][j] = 0.f;

    for (int h = 0; h < 64; ++h) {
      v8f s[4];
      for (int nil = 0; nil < 4; ++nil)
        for (int j = 0; j < 8; ++j) s[nil][j] = 0.f;
#pragma unroll
      for (int ks = 0; ks < 4; ++ks) {
        ABf af;
        const __bf16* ap = qrow + h * HD + ks * 32 + dA;
        af.h[0] = *(const v8bf*)ap;
        af.h[1] = *(const v8bf*)(ap + 16);
#pragma unroll
        for (int nil = 0; nil < 4; ++nil)
          s[nil] = wmma_bf16(af.v, bfr[nil][ks].v, s[nil]);
      }
      // per-head weights: 8 consecutive floats (broadcast within lane half)
      const v4f* wp = (const v4f*)&wt[h * 128 + wv * 16 + rj];
      v4f w0 = wp[0], w1 = wp[1];
#pragma unroll
      for (int nil = 0; nil < 4; ++nil) {
#pragma unroll
        for (int j = 0; j < 8; ++j) {
          float sv = s[nil][j];
          sv = sv > 0.f ? sv : 0.f;                   // relu
          float wvv = (j < 4) ? w0[j] : w1[j - 4];    // w[q_row, h]
          acc[nil][j] += wvv * sv;
        }
      }
    }
    // Epilogue for these 4 key sub-tiles
#pragma unroll
    for (int nil = 0; nil < 4; ++nil) {
      int keyg = kbase + (npass * 4 + nil) * 16 + m;
#pragma unroll
      for (int j = 0; j < 8; ++j) {
        int qpos = qbase + wv * 16 + rj + j;
        float v = acc[nil][j] * SCALE_F;
        if (keyg > qpos) v = -__builtin_inff();  // causal mask
        scores[(size_t)qpos * T_TOK + keyg] = v;
      }
    }
  }
}

// ---------------------------------------------------------------------------
// Kernel 4: per-row top-2048 (sorted desc, with indices) via bitonic sort of
// 4096 (val,idx) pairs in 32 KB LDS. One block per query row.
// ---------------------------------------------------------------------------
__global__ __launch_bounds__(256) void mla_topk(
    const float* __restrict__ scores, float* __restrict__ outv,
    int* __restrict__ outi) {
  __shared__ float sv[4096];
  __shared__ int si[4096];
  const int t = threadIdx.x, q = blockIdx.x;
  for (int i = 0; i < 16; ++i) {
    int idx = t + 256 * i;
    sv[idx] = scores[(size_t)q * T_TOK + idx];
    si[idx] = idx;
  }
  __syncthreads();
  for (int k = 2; k <= 4096; k <<= 1) {
    for (int j = k >> 1; j > 0; j >>= 1) {
      for (int p = t; p < 2048; p += 256) {
        int i1 = ((p & ~(j - 1)) << 1) | (p & (j - 1));
        int i2 = i1 | j;
        bool dirDesc = ((i1 & k) == 0);
        float a = sv[i1], b = sv[i2];
        bool sw = dirDesc ? (a < b) : (a > b);
        if (sw) {
          sv[i1] = b; sv[i2] = a;
          int ti = si[i1]; si[i1] = si[i2]; si[i2] = ti;
        }
      }
      __syncthreads();
    }
  }
  for (int i = 0; i < 8; ++i) {
    int idx = t + 256 * i;
    outv[(size_t)q * TOPK_N + idx] = sv[idx];
    outi[(size_t)q * TOPK_N + idx] = si[idx];
  }
}

// ---------------------------------------------------------------------------
extern "C" void kernel_launch(void* const* d_in, const int* in_sizes, int n_in,
                              void* d_out, int out_size, void* d_ws,
                              size_t ws_size, hipStream_t stream) {
  (void)in_sizes; (void)n_in; (void)out_size; (void)ws_size;
  const float* x    = (const float*)d_in[0];
  const float* qr   = (const float*)d_in[1];
  const float* cosp = (const float*)d_in[2];
  const float* sinp = (const float*)d_in[3];
  const float* wqb  = (const float*)d_in[4];
  const float* wk   = (const float*)d_in[5];
  const float* ww   = (const float*)d_in[6];
  const float* knw  = (const float*)d_in[7];
  const float* knb  = (const float*)d_in[8];

  char* ws = (char*)d_ws;
  __bf16* qb     = (__bf16*)ws;                           // 64 MB
  __bf16* kb     = (__bf16*)(ws + ((size_t)64 << 20));    // 1 MB
  float*  wbuf   = (float*)(ws + ((size_t)65 << 20));     // 1 MB
  float*  scores = (float*)(ws + ((size_t)66 << 20));     // 64 MB

  float* outv = (float*)d_out;
  int*   outi = (int*)d_out + (size_t)T_TOK * TOPK_N;

  mla_qgemm_rope<<<dim3(QCOLS / 128, T_TOK / 128), 256, 0, stream>>>(
      qr, wqb, cosp, sinp, qb);
  mla_kwprep<<<T_TOK, 256, 0, stream>>>(x, wk, ww, knw, knb, cosp, sinp, kb,
                                        wbuf);
  mla_score<<<dim3(T_TOK / 128, T_TOK / 128), 256, 0, stream>>>(qb, kb, wbuf,
                                                                scores);
  mla_topk<<<T_TOK, 256, 0, stream>>>(scores, outv, outi);
}
